// SwinBlock_89318139888127
// MI455X (gfx1250) — matrix operand order repaired and checked by reading, not compiled, not run on hardware
//
#include <hip/hip_runtime.h>

// ---------------------------------------------------------------------------
// 3D Swin block for MI455X (gfx1250, wave32, WMMA).
// One workgroup (256 threads = 8 waves) per 4x4x4 window (4096 windows).
// All per-window intermediates live in LDS; every GEMM uses
// v_wmma_f32_16x16x32_f16 (f16 operands, f32 accumulate).
// Weights are packed to f16 once per launch; B-fragments are pure
// global_load_b128 pairs.  K-tiles are preloaded into distinct registers
// before the WMMA chain so matrix issue is not serialized by load waits.
// ---------------------------------------------------------------------------

typedef __attribute__((ext_vector_type(16))) _Float16 v16h;
typedef __attribute__((ext_vector_type(8)))  float    v8f;

#define TPB   256
#define WAVES 8

// LDS layout (bytes). Regions are reused across phases:
//   XH region : ln1-out -> attn-out O -> mlp-ln Hh
//   QKV+VT    : qkv / k / transposed-v -> MLP hidden G (64x384 f16 = 49152B)
//   S region  : (pass2: rolled gather buffer Xs) -> attention scores f32
#define OFF_XF32   0         // 64*96*4   = 24576  residual accumulator (f32)
#define OFF_XH     24576     // 64*96*2   = 12288
#define OFF_QKV    36864     // 64*288*2  = 36864
#define OFF_VT     73728     // 96*64*2   = 12288
#define OFF_S      86016     // 3*64*64*4 = 49152
#define OFF_PH     135168    // 3*64*64*2 = 24576
#define LDS_BYTES  159744

// Packed f16 weight arena (halves, relative to arena base in d_ws):
//   layer1: qkv @0 (27648), proj @27648 (9216), w1 @36864 (36864), w2 @73728 (36864)
//   layer2: same shapes at +110592
#define WQKV_N   27648
#define WPROJ_N  9216
#define W1_N     36864
#define W2_N     36864
#define WLAYER_N 110592
#define WTOT_N   221184

// ---- WMMA fragment helpers (wave32; A/B use contiguous k order) -------------------

// A (16x32 f16): lane&15 = row M; K = 16*(lane>>4) + 0..15 (contiguous).
// Source: row-major f16 in LDS, leading dim `ld` halves.
__device__ __forceinline__ v16h load_a_lds(const _Float16* base, int ld,
                                           int row0, int k0, int lane) {
  const _Float16* p = base + (row0 + (lane & 15)) * ld + k0 + ((lane >> 4) & 1) * 16;
  v16h a;
#pragma unroll
  for (int i = 0; i < 16; ++i) a[i] = p[i];
  return a;
}

// B (32x16 f16): lane&15 = col N; K = 16*(lane>>4) + 0..15 (contiguous).
// B[k][n] = W[n][k] for Y = X @ W^T with W row-major (O x C).
__device__ __forceinline__ v16h load_b_lds(const _Float16* base, int ld,
                                           int k0, int n0, int lane) {
  const _Float16* p = base + (n0 + (lane & 15)) * ld + k0 + ((lane >> 4) & 1) * 16;
  v16h b;
#pragma unroll
  for (int i = 0; i < 16; ++i) b[i] = p[i];
  return b;
}

// Same layout, but from packed f16 weights in global: 16 consecutive halves
// per lane = two global_load_b128, zero conversion ALU.
__device__ __forceinline__ v16h load_b_gh(const _Float16* __restrict__ W, int ld,
                                          int k0, int n0, int lane) {
  const _Float16* p = W + (n0 + (lane & 15)) * ld + k0 + ((lane >> 4) & 1) * 16;
  v16h b;
#pragma unroll
  for (int i = 0; i < 16; ++i) b[i] = p[i];
  return b;
}

__device__ __forceinline__ v8f wmma16(v16h a, v16h b, v8f c) {
  return __builtin_amdgcn_wmma_f32_16x16x32_f16(false, a, false, b,
                                                (short)0, c, false, false);
}

// ---------------------------------------------------------------------------
// One-shot weight pack: all 8 weight matrices f32 -> f16 arena in d_ws.
// ---------------------------------------------------------------------------
__global__ __launch_bounds__(TPB)
void pack_weights_kernel(const float* __restrict__ q1, const float* __restrict__ p1,
                         const float* __restrict__ w11, const float* __restrict__ w12,
                         const float* __restrict__ q2, const float* __restrict__ p2,
                         const float* __restrict__ w21, const float* __restrict__ w22,
                         _Float16* __restrict__ out) {
  const int i = blockIdx.x * TPB + threadIdx.x;
  if (i >= WTOT_N) return;
  const float* s;
  int off;
  if      (i < 27648)  { s = q1;  off = 0;      }
  else if (i < 36864)  { s = p1;  off = 27648;  }
  else if (i < 73728)  { s = w11; off = 36864;  }
  else if (i < 110592) { s = w12; off = 73728;  }
  else if (i < 138240) { s = q2;  off = 110592; }
  else if (i < 147456) { s = p2;  off = 138240; }
  else if (i < 184320) { s = w21; off = 147456; }
  else                 { s = w22; off = 184320; }
  out[i] = (_Float16)s[i - off];
}

// ---------------------------------------------------------------------------
// SECOND=false: pass 1 (un-shifted windows): src = x, dst = mid.
// SECOND=true : pass 2 (shifted windows):    src = mid, dst = out; applies
//               roll(-2,-2,-2), cross-layout residual, and the window mask.
// ---------------------------------------------------------------------------
template <bool SECOND>
__global__ __launch_bounds__(TPB, 1)
void swin_block_kernel(const float* __restrict__ src,
                       const float* __restrict__ mask,
                       const float* __restrict__ ng, const float* __restrict__ nb,
                       const _Float16* __restrict__ wbase_h,  // packed layer weights
                       const float* __restrict__ proj_b,
                       const float* __restrict__ tab,
                       const float* __restrict__ mg, const float* __restrict__ mb,
                       const float* __restrict__ b1, const float* __restrict__ b2,
                       float* __restrict__ dst) {
  extern __shared__ char smem[];
  float*    Xf = (float*)(smem + OFF_XF32);
  _Float16* Xh = (_Float16*)(smem + OFF_XH);
  _Float16* QK = (_Float16*)(smem + OFF_QKV);
  _Float16* VT = (_Float16*)(smem + OFF_VT);
  float*    S  = (float*)(smem + OFF_S);
  _Float16* Ph = (_Float16*)(smem + OFF_PH);
  _Float16* G  = (_Float16*)(smem + OFF_QKV);  // MLP hidden, reuses QKV+VT

  const _Float16* qkv_w  = wbase_h;
  const _Float16* proj_w = wbase_h + WQKV_N;
  const _Float16* w1     = wbase_h + WQKV_N + WPROJ_N;
  const _Float16* w2     = wbase_h + WQKV_N + WPROJ_N + W1_N;

  const int tid  = threadIdx.x;
  const int lane = tid & 31;
  const int wave = tid >> 5;
  const int widx = blockIdx.x;                 // (dq*16 + hq)*16 + wq
  const int dq = widx >> 8, hq = (widx >> 4) & 15, wq = widx & 15;
  const int wbase = dq * 16384 + hq * 256 + wq * 4;

  // ---- gather window (raw-view _win_part indexing), plus pass-2 residual --
  for (int idx = tid; idx < 6144; idx += TPB) {
    const int cc = idx % 96;                 // "channel" of raw view
    const int t  = idx / 96;                 // token = dr*16 + hr*4 + wr
    const int dr = t >> 4, hr = (t >> 2) & 3, wr = t & 3;
    const int flat = cc * 262144 + wbase + dr * 4096 + hr * 64 + wr;
    if (!SECOND) {
      Xf[idx] = src[flat];
    } else {
      // decode raw flat index into normal (d,h,w,c), apply roll(-2,-2,-2)
      int c = flat % 96;
      int rem = flat / 96;
      int w = rem & 63; rem >>= 6;
      int h = rem & 63;
      int d = rem >> 6;
      S[idx] = src[((((d + 2) & 63) * 64 + ((h + 2) & 63)) * 64 +
                    ((w + 2) & 63)) * 96 + c];
      // residual is taken at the UN-shifted window position (reference quirk)
      const int dd = dq * 4 + dr, hh = hq * 4 + hr, ww = wq * 4 + wr;
      Xf[idx] = src[((dd * 64 + hh) * 64 + ww) * 96 + cc];
    }
  }
  __syncthreads();

  // ---- LayerNorm (attention pre-norm) -> Xh (f16) -------------------------
  {
    const float* li = SECOND ? S : Xf;
    if (tid < 64) {
      const float* r = li + tid * 96;
      float m = 0.f;
#pragma unroll 8
      for (int c = 0; c < 96; ++c) m += r[c];
      m *= (1.f / 96.f);
      float v = 0.f;
#pragma unroll 8
      for (int c = 0; c < 96; ++c) { float dlt = r[c] - m; v += dlt * dlt; }
      v *= (1.f / 96.f);
      const float rs = rsqrtf(v + 1e-5f);
      for (int c = 0; c < 96; ++c)
        Xh[tid * 96 + c] = (_Float16)((r[c] - m) * rs * ng[c] + nb[c]);
    }
  }
  __syncthreads();

  // ---- QKV: Y[64x288] = Xh @ qkv_w^T  (q pre-scaled; v also transposed) ---
  for (int tt = wave; tt < 72; tt += WAVES) {
    const int mt = tt / 18, nt = tt % 18;
    // q/k/v segment is uniform per 16-wide N tile (96 % 16 == 0)
    const bool isQ = (nt < 6);
    const bool isV = (nt >= 12);
    v16h af[3], bf[3];
#pragma unroll
    for (int kt = 0; kt < 3; ++kt) {
      af[kt] = load_a_lds(Xh, 96, mt * 16, kt * 32, lane);
      bf[kt] = load_b_gh(qkv_w, 96, kt * 32, nt * 16, lane);
    }
    v8f acc = {};
#pragma unroll
    for (int kt = 0; kt < 3; ++kt) acc = wmma16(af[kt], bf[kt], acc);
    const int g = lane >> 4, nl = lane & 15;
    const int col = nt * 16 + nl;
    if (isQ) {
#pragma unroll
      for (int v = 0; v < 8; ++v) {
        const int row = mt * 16 + 8 * g + v;
        QK[row * 288 + col] = (_Float16)(acc[v] * 0.17677669529663687f);
      }
    } else if (isV) {
#pragma unroll
      for (int v = 0; v < 8; ++v) {
        const int row = mt * 16 + 8 * g + v;
        const _Float16 hv = (_Float16)acc[v];
        QK[row * 288 + col] = hv;
        VT[(col - 192) * 64 + row] = hv;                 // V^T for AV B-frags
      }
    } else {
#pragma unroll
      for (int v = 0; v < 8; ++v) {
        const int row = mt * 16 + 8 * g + v;
        QK[row * 288 + col] = (_Float16)acc[v];
      }
    }
  }
  __syncthreads();

  // ---- scores: S[h] = q_h k_h^T + rel-pos bias (+ mask in pass 2) ---------
  for (int tt = wave; tt < 48; tt += WAVES) {
    const int h = tt >> 4, mt = (tt >> 2) & 3, nt = tt & 3;
    v16h a = load_a_lds(QK, 288, mt * 16, h * 32, lane);
    v16h b = load_b_lds(QK + 96 + h * 32, 288, 0, nt * 16, lane);
    v8f acc = {};
    acc = wmma16(a, b, acc);
    const int g = lane >> 4, nl = lane & 15;
    const int j = nt * 16 + nl;
    const int dj = j >> 4, hj = (j >> 2) & 3, wj = j & 3;
#pragma unroll
    for (int v = 0; v < 8; ++v) {
      const int i = mt * 16 + 8 * g + v;
      const int di = i >> 4, hi = (i >> 2) & 3, wi = i & 3;
      const int rel = ((di - dj + 3) * 7 + (hi - hj + 3)) * 7 + (wi - wj + 3);
      float val = acc[v] + tab[rel * 3 + h];
      if (SECOND) val += mask[(size_t)(widx * 64 + i) * 64 + j];
      S[h * 4096 + i * 64 + j] = val;
    }
  }
  __syncthreads();

  // ---- softmax rows -> Ph (f16 probabilities) -----------------------------
  if (tid < 192) {
    float*    r  = S  + (tid >> 6) * 4096 + (tid & 63) * 64;
    _Float16* pr = Ph + (tid >> 6) * 4096 + (tid & 63) * 64;
    float mx = r[0];
    for (int j = 1; j < 64; ++j) mx = fmaxf(mx, r[j]);
    float sum = 0.f;
    for (int j = 0; j < 64; ++j) { float e = expf(r[j] - mx); r[j] = e; sum += e; }
    const float inv = 1.f / sum;
    for (int j = 0; j < 64; ++j) pr[j] = (_Float16)(r[j] * inv);
  }
  __syncthreads();

  // ---- AV: O_h = P_h @ V_h  (B fragments from V^T, contiguous) ------------
  _Float16* O = Xh;  // reuse ln buffer for the 64x96 attention output
  for (int tt = wave; tt < 24; tt += WAVES) {
    const int h = tt >> 3, mt = (tt >> 1) & 3, nt = tt & 1;
    v16h af[2], bf[2];
#pragma unroll
    for (int kt = 0; kt < 2; ++kt) {
      af[kt] = load_a_lds(Ph + h * 4096, 64, mt * 16, kt * 32, lane);
      bf[kt] = load_b_lds(VT + h * 2048, 64, kt * 32, nt * 16, lane);
    }
    v8f acc = {};
#pragma unroll
    for (int kt = 0; kt < 2; ++kt) acc = wmma16(af[kt], bf[kt], acc);
    const int g = lane >> 4, nl = lane & 15;
    const int col = h * 32 + nt * 16 + nl;
#pragma unroll
    for (int v = 0; v < 8; ++v)
      O[(mt * 16 + 8 * g + v) * 96 + col] = (_Float16)acc[v];
  }
  __syncthreads();

  // ---- proj: Xf += O @ proj_w^T + proj_b ----------------------------------
  for (int tt = wave; tt < 24; tt += WAVES) {
    const int mt = tt / 6, nt = tt % 6;
    v16h af[3], bf[3];
#pragma unroll
    for (int kt = 0; kt < 3; ++kt) {
      af[kt] = load_a_lds(O, 96, mt * 16, kt * 32, lane);
      bf[kt] = load_b_gh(proj_w, 96, kt * 32, nt * 16, lane);
    }
    v8f acc = {};
#pragma unroll
    for (int kt = 0; kt < 3; ++kt) acc = wmma16(af[kt], bf[kt], acc);
    const int g = lane >> 4, nl = lane & 15;
    const int col = nt * 16 + nl;
    const float pb = proj_b[col];
#pragma unroll
    for (int v = 0; v < 8; ++v)
      Xf[(mt * 16 + 8 * g + v) * 96 + col] += acc[v] + pb;
  }
  __syncthreads();

  // ---- MLP pre-norm -> Hh (reuses Xh region; O is dead) -------------------
  if (tid < 64) {
    const float* r = Xf + tid * 96;
    float m = 0.f;
#pragma unroll 8
    for (int c = 0; c < 96; ++c) m += r[c];
    m *= (1.f / 96.f);
    float v = 0.f;
#pragma unroll 8
    for (int c = 0; c < 96; ++c) { float dlt = r[c] - m; v += dlt * dlt; }
    v *= (1.f / 96.f);
    const float rs = rsqrtf(v + 1e-5f);
    for (int c = 0; c < 96; ++c)
      Xh[tid * 96 + c] = (_Float16)((r[c] - m) * rs * mg[c] + mb[c]);
  }
  __syncthreads();

  // ---- MLP fc1 + exact GELU -> G[64x384] ----------------------------------
  for (int tt = wave; tt < 96; tt += WAVES) {
    const int mt = tt / 24, nt = tt % 24;
    v16h af[3], bf[3];
#pragma unroll
    for (int kt = 0; kt < 3; ++kt) {
      af[kt] = load_a_lds(Xh, 96, mt * 16, kt * 32, lane);
      bf[kt] = load_b_gh(w1, 96, kt * 32, nt * 16, lane);
    }
    v8f acc = {};
#pragma unroll
    for (int kt = 0; kt < 3; ++kt) acc = wmma16(af[kt], bf[kt], acc);
    const int g = lane >> 4, nl = lane & 15;
    const int col = nt * 16 + nl;
    const float bb = b1[col];
#pragma unroll
    for (int v = 0; v < 8; ++v) {
      const float x = acc[v] + bb;
      const float ge = 0.5f * x * (1.f + erff(x * 0.70710678118654752f));
      G[(mt * 16 + 8 * g + v) * 384 + col] = (_Float16)ge;
    }
  }
  __syncthreads();

  // ---- MLP fc2: Xf += G @ w2^T + b2  (12 k-tiles, preload in chunks of 4) -
  for (int tt = wave; tt < 24; tt += WAVES) {
    const int mt = tt / 6, nt = tt % 6;
    v8f acc = {};
#pragma unroll
    for (int kc = 0; kc < 3; ++kc) {
      v16h af[4], bf[4];
#pragma unroll
      for (int u = 0; u < 4; ++u) {
        const int kt = kc * 4 + u;
        af[u] = load_a_lds(G, 384, mt * 16, kt * 32, lane);
        bf[u] = load_b_gh(w2, 384, kt * 32, nt * 16, lane);
      }
#pragma unroll
      for (int u = 0; u < 4; ++u) acc = wmma16(af[u], bf[u], acc);
    }
    const int g = lane >> 4, nl = lane & 15;
    const int col = nt * 16 + nl;
    const float bb = b2[col];
#pragma unroll
    for (int v = 0; v < 8; ++v)
      Xf[(mt * 16 + 8 * g + v) * 96 + col] += acc[v] + bb;
  }
  __syncthreads();

  // ---- window-reverse store (normal (D,H,W,C) layout) ---------------------
  for (int idx = tid; idx < 6144; idx += TPB) {
    const int cc = idx % 96;
    const int t  = idx / 96;
    const int dr = t >> 4, hr = (t >> 2) & 3, wr = t & 3;
    const int dd = dq * 4 + dr, hh = hq * 4 + hr, ww = wq * 4 + wr;
    dst[((dd * 64 + hh) * 64 + ww) * 96 + cc] = Xf[idx];
  }
}

// ---------------------------------------------------------------------------

extern "C" void kernel_launch(void* const* d_in, const int* in_sizes, int n_in,
                              void* d_out, int out_size, void* d_ws, size_t ws_size,
                              hipStream_t stream) {
  (void)in_sizes; (void)n_in; (void)out_size; (void)ws_size;

  const float* x      = (const float*)d_in[0];
  const float* mask   = (const float*)d_in[1];
  const float* n1g    = (const float*)d_in[2];
  const float* n1b    = (const float*)d_in[3];
  const float* n2g    = (const float*)d_in[4];
  const float* n2b    = (const float*)d_in[5];
  const float* a1_qkv = (const float*)d_in[6];
  const float* a1_pw  = (const float*)d_in[7];
  const float* a1_pb  = (const float*)d_in[8];
  const float* a1_tab = (const float*)d_in[9];
  const float* a2_qkv = (const float*)d_in[10];
  const float* a2_pw  = (const float*)d_in[11];
  const float* a2_pb  = (const float*)d_in[12];
  const float* a2_tab = (const float*)d_in[13];
  const float* m1g    = (const float*)d_in[14];
  const float* m1b    = (const float*)d_in[15];
  const float* m1w1   = (const float*)d_in[16];
  const float* m1b1   = (const float*)d_in[17];
  const float* m1w2   = (const float*)d_in[18];
  const float* m1b2   = (const float*)d_in[19];
  const float* m2g    = (const float*)d_in[20];
  const float* m2b    = (const float*)d_in[21];
  const float* m2w1   = (const float*)d_in[22];
  const float* m2b1   = (const float*)d_in[23];
  const float* m2w2   = (const float*)d_in[24];
  const float* m2b2   = (const float*)d_in[25];

  float* out = (float*)d_out;
  float* mid = (float*)d_ws;   // 64*64*64*96 f32 intermediate (~100 MB)
  _Float16* wpack = (_Float16*)((char*)d_ws + (size_t)25165824 * sizeof(float));

  // Pass 0: pack all weights to f16 (L2-resident, reused by 8192 blocks)
  pack_weights_kernel<<<dim3((WTOT_N + TPB - 1) / TPB), dim3(TPB), 0, stream>>>(
      a1_qkv, a1_pw, m1w1, m1w2, a2_qkv, a2_pw, m2w1, m2w2, wpack);

  dim3 grid(4096), block(TPB);

  // Pass 1: un-shifted windows, x -> mid
  swin_block_kernel<false><<<grid, block, LDS_BYTES, stream>>>(
      x, mask, n1g, n1b, wpack, a1_pb, a1_tab,
      m1g, m1b, m1b1, m1b2, mid);

  // Pass 2: shifted windows (roll -2,-2,-2), mid -> out
  swin_block_kernel<true><<<grid, block, LDS_BYTES, stream>>>(
      mid, mask, n2g, n2b, wpack + WLAYER_N, a2_pb, a2_tab,
      m2g, m2b, m2b1, m2b2, out);
}
